// MHA_33122787787429
// MI455X (gfx1250) — compile-verified, matrix operand order checked by reading, not run on hardware
//
#include <hip/hip_runtime.h>

typedef __attribute__((ext_vector_type(16))) __bf16 v16bf;
typedef __attribute__((ext_vector_type(2)))  __bf16 v2bf;
typedef __attribute__((ext_vector_type(8)))  float  v8f;
typedef __attribute__((ext_vector_type(4)))  float  v4f;
typedef unsigned int u32x4 __attribute__((ext_vector_type(4)));
typedef int          i32x4 __attribute__((ext_vector_type(4)));
typedef int          i32x8 __attribute__((ext_vector_type(8)));

#define DM    768
#define NSEQ  4096
#define BATCH 2
#define NH    12
#define DK    64
#define MROWS (BATCH * NSEQ)   // 8192
#define NEGBIG (-1.0e9f)
#define PSE   72               // LDS P row stride (elements)
#define KVS   72               // LDS K/V row stride (elements) = 128B + 16B TDM pad
#define WROW  792              // LDS W row stride (elements)   = 1536B + 3x16B pads

// Interleaved position of offset within a 32-wide contraction block: pairs
// (k, k+16) adjacent -> every WMMA operand load is one contiguous 32B read;
// A and B share the same K-permutation so the contraction is unchanged.
static __device__ __forceinline__ int ilv(int off) {
  return (off < 16) ? (off * 2) : ((off - 16) * 2 + 1);
}

static __device__ __forceinline__ v8f wmma_bf16(v16bf a, v16bf b, v8f c) {
  return __builtin_amdgcn_wmma_f32_16x16x32_bf16(false, a, false, b, (short)0, c,
                                                 false, false);
}

static __device__ __forceinline__ unsigned lds_off(const void* p) {
  return (unsigned)(uintptr_t)p;   // low 32 bits of LDS aperture addr = byte offset
}

// ---- Tensor Data Mover: 2-D tile (bf16) global -> LDS with stream padding.
// pad_code: pad inserted every 2^(pad_code+1) dwords of stored data;
// pad amount fixed at 4 dwords (16B).
static __device__ __forceinline__ void tdm_load_2d(unsigned ldsoff,
                                                   const void* gptr,
                                                   unsigned tile_d0,
                                                   unsigned tile_d1,
                                                   unsigned stride_el,
                                                   unsigned pad_code) {
  unsigned long long ga = (unsigned long long)(uintptr_t)gptr;
  u32x4 g0;
  g0[0] = 1u;                                    // count=1, user descriptor
  g0[1] = ldsoff;                                // lds_addr (bytes)
  g0[2] = (unsigned)ga;                          // global_addr[31:0]
  g0[3] = (unsigned)((ga >> 32) & 0x01FFFFFFu) | (2u << 30);  // addr[56:32]|type=2

  i32x8 g1;
  g1[0] = (int)((1u << 16) | (1u << 20) | (pad_code << 22) | (3u << 25));
          // data_size=2B, pad_enable, pad_interval, pad_amount=4 dwords
  g1[1] = (int)(tile_d0 << 16);                  // tensor_dim0[15:0]
  g1[2] = (int)(tile_d1 << 16);                  // tensor_dim1[15:0]
  g1[3] = (int)(tile_d0 << 16);                  // tile_dim0
  g1[4] = (int)tile_d1;                          // tile_dim1 (tile_dim2=0)
  g1[5] = (int)stride_el;                        // tensor_dim0_stride[31:0]
  g1[6] = 0;
  g1[7] = 0;

  i32x4 z4 = {0, 0, 0, 0};
#if defined(__clang_major__) && __clang_major__ >= 23
  i32x8 z8 = {0, 0, 0, 0, 0, 0, 0, 0};
  __builtin_amdgcn_tensor_load_to_lds(g0, g1, z4, z4, z8, 0);
#else
  __builtin_amdgcn_tensor_load_to_lds(g0, g1, z4, z4, 0);
#endif
}

// fp32 A-tile -> bf16 register tile in interleaved-K element order.
static __device__ __forceinline__ v16bf load_f32_perm(const float* p) {
  v4f a = *(const v4f*)(p);
  v4f b = *(const v4f*)(p + 4);
  v4f c = *(const v4f*)(p + 16);
  v4f d = *(const v4f*)(p + 20);
  v16bf r;
  #pragma unroll
  for (int i = 0; i < 4; ++i) {
    r[2 * i]           = (__bf16)a[i];
    r[2 * i + 1]       = (__bf16)c[i];
    r[2 * (i + 4)]     = (__bf16)b[i];
    r[2 * (i + 4) + 1] = (__bf16)d[i];
  }
  return r;
}

// ---------------------------------------------------------------- weights cvt
__global__ void cvt_k(const float* __restrict__ src, __bf16* __restrict__ dst,
                      int n, float scale) {
  int i = blockIdx.x * blockDim.x + threadIdx.x;
  if (i < n) {
    int row = i / DM, col = i % DM;
    dst[(size_t)row * DM + (col & ~31) + ilv(col & 31)] = (__bf16)(src[i] * scale);
  }
}

// ---------------------------------------------------------------- GEMM
// C[m,o] = sum_d A[m,d] * W[o,d].  The whole 64x768 W block is staged once
// via one TDM descriptor (one barrier total), then 24 K-chunks run sync-free.
template <int MODE>
__global__ __launch_bounds__(256) void gemm_k(const void* __restrict__ Ax,
                                              const __bf16* __restrict__ W,
                                              void* __restrict__ Out) {
  __shared__ __align__(16) __bf16 lsW[64 * WROW];   // ~99 KB

  const int lane = threadIdx.x & 31;
  const int wave = threadIdx.x >> 5;
  const int hi   = lane >> 4;
  const int nl   = lane & 15;
  const int m0   = blockIdx.x * 128 + wave * 16;
  const int n0   = blockIdx.y * 64;

  const float*  Af = (const float*)Ax;
  const __bf16* Ab = (const __bf16*)Ax;
  const int arow = m0 + nl;

  if (wave == 0)   // 64 rows x 768 elems, pad 16B per 512B stored
    tdm_load_2d(lds_off(&lsW[0]), W + (size_t)n0 * DM, DM, 64, DM, 6);

  v8f acc[4] = {};

  __builtin_amdgcn_s_wait_tensorcnt(0);
  __syncthreads();

  for (int c = 0; c < DM / 32; ++c) {
    v16bf a;
    if (MODE < 3) a = load_f32_perm(Af + (size_t)arow * DM + c * 32 + hi * 8);
    else          a = *(const v16bf*)(Ab + (size_t)arow * DM + c * 32 + hi * 16);

    const int coff = c * 32 + 8 * (c >> 3);   // chunk offset incl. TDM pads (elems)
    #pragma unroll
    for (int t = 0; t < 4; ++t) {
      v16bf b = *(const v16bf*)(&lsW[(16 * t + nl) * WROW + coff + hi * 16]);
      acc[t] = wmma_bf16(a, b, acc[t]);
    }
  }

  #pragma unroll
  for (int t = 0; t < 4; ++t) {
    const int o = n0 + 16 * t + nl;
    #pragma unroll
    for (int r = 0; r < 8; ++r) {
      const int m = m0 + r + hi * 8;
      const float v = acc[t][r];
      if (MODE == 3) {
        ((float*)Out)[(size_t)m * DM + o] = v;
      } else {
        const int b2 = m >> 12, nidx = m & (NSEQ - 1);
        const int h = o >> 6, d = o & (DK - 1);
        if (MODE == 2)
          ((__bf16*)Out)[((size_t)(b2 * NH + h) * DK + d) * NSEQ +
                         (nidx & ~31) + ilv(nidx & 31)] = (__bf16)v;
        else
          ((__bf16*)Out)[((size_t)(b2 * NH + h) * NSEQ + nidx) * DK +
                         (d & 32) + ilv(d & 31)] = (__bf16)v;
      }
    }
  }
}

// ---------------------------------------------------------------- flash attn
// 16-query tile per wave, 64-key chunks staged cooperatively via TDM
// double-buffer; 128 WMMAs per WG between barriers.
__global__ __launch_bounds__(256) void flash_k(const __bf16* __restrict__ qp,
                                               const __bf16* __restrict__ kp,
                                               const __bf16* __restrict__ vt,
                                               __bf16* __restrict__ attn) {
  __shared__ __align__(16) __bf16 lsK[2][64 * KVS];
  __shared__ __align__(16) __bf16 lsV[2][64 * KVS];
  __shared__ __align__(16) __bf16 lds_p[8][16 * PSE];

  const int lane = threadIdx.x & 31;
  const int wave = threadIdx.x >> 5;
  const int hi   = lane >> 4;
  const int nl   = lane & 15;

  const int bh = blockIdx.y;
  const int b2 = bh / NH, h = bh % NH;
  const int q0 = (blockIdx.x * 8 + wave) * 16;

  const __bf16* qh = qp + (size_t)bh * NSEQ * DK;
  const __bf16* kh = kp + (size_t)bh * NSEQ * DK;
  const __bf16* vh = vt + (size_t)bh * DK * NSEQ;
  __bf16* lp = &lds_p[wave][0];

  const v16bf aq0 = *(const v16bf*)(qh + (size_t)(q0 + nl) * DK + 0  + hi * 16);
  const v16bf aq1 = *(const v16bf*)(qh + (size_t)(q0 + nl) * DK + 32 + hi * 16);

  v8f acc[4] = {};
  float mrow[8], lrow[8];
  #pragma unroll
  for (int r = 0; r < 8; ++r) { mrow[r] = -3.0e38f; lrow[r] = 0.0f; }

  const int nch_wg = 2 * blockIdx.x + 2;      // WG-uniform chunk count

  if (wave == 0) {
    tdm_load_2d(lds_off(&lsK[0][0]), kh, 64, 64, DK, 4);
    tdm_load_2d(lds_off(&lsV[0][0]), vh, 64, 64, NSEQ, 4);
  }

  for (int c = 0; c < nch_wg; ++c) {
    const int k0 = c << 6;
    __builtin_amdgcn_s_wait_tensorcnt(0);
    __syncthreads();
    if (wave == 0 && c + 1 < nch_wg) {
      tdm_load_2d(lds_off(&lsK[(c + 1) & 1][0]),
                  kh + (size_t)(k0 + 64) * DK, 64, 64, DK, 4);
      tdm_load_2d(lds_off(&lsV[(c + 1) & 1][0]),
                  vh + (k0 + 64), 64, 64, NSEQ, 4);
    }
    if (k0 > q0 + 15) continue;                // causally done for this wave

    const __bf16* lbK = &lsK[c & 1][0];
    const __bf16* lbV = &lsV[c & 1][0];

    // ---- scores: 4 key half-tiles x 2 dk slices = 8 WMMAs (B from LDS)
    v8f s[4] = {};
    #pragma unroll
    for (int dc = 0; dc < 2; ++dc) {
      const v16bf aq = dc ? aq1 : aq0;
      #pragma unroll
      for (int t = 0; t < 4; ++t) {
        v16bf bk = *(const v16bf*)(lbK + (16 * t + nl) * KVS + dc * 32 + hi * 16);
        s[t] = wmma_bf16(aq, bk, s[t]);
      }
    }

    const bool diag = (k0 + 63) > q0;

    // ---- online softmax
    #pragma unroll
    for (int r = 0; r < 8; ++r) {
      const int q = q0 + r + hi * 8;
      float x0 = s[0][r], x1 = s[1][r], x2 = s[2][r], x3 = s[3][r];
      if (diag) {
        if (k0 + nl > q)      x0 = NEGBIG;
        if (k0 + 16 + nl > q) x1 = NEGBIG;
        if (k0 + 32 + nl > q) x2 = NEGBIG;
        if (k0 + 48 + nl > q) x3 = NEGBIG;
      }
      float mc = fmaxf(fmaxf(x0, x1), fmaxf(x2, x3));
      mc = fmaxf(mc, __shfl_xor(mc, 1, 32));
      mc = fmaxf(mc, __shfl_xor(mc, 2, 32));
      mc = fmaxf(mc, __shfl_xor(mc, 4, 32));
      mc = fmaxf(mc, __shfl_xor(mc, 8, 32));
      const float mn = fmaxf(mrow[r], mc);
      const float sc = __expf(mrow[r] - mn);
      const float p0 = __expf(x0 - mn);
      const float p1 = __expf(x1 - mn);
      const float p2 = __expf(x2 - mn);
      const float p3 = __expf(x3 - mn);
      float sm = (p0 + p1) + (p2 + p3);
      sm += __shfl_xor(sm, 1, 32);
      sm += __shfl_xor(sm, 2, 32);
      sm += __shfl_xor(sm, 4, 32);
      sm += __shfl_xor(sm, 8, 32);
      lrow[r] = lrow[r] * sc + sm;
      mrow[r] = mn;
      acc[0][r] *= sc; acc[1][r] *= sc; acc[2][r] *= sc; acc[3][r] *= sc;

      const int m = r + hi * 8;
      v2bf pa; pa[0] = (__bf16)p0; pa[1] = (__bf16)p1;
      v2bf pb; pb[0] = (__bf16)p2; pb[1] = (__bf16)p3;
      *(v2bf*)(lp + m * PSE + 2 * nl)      = pa;
      *(v2bf*)(lp + m * PSE + 32 + 2 * nl) = pb;
    }
    asm volatile("" ::: "memory");   // DS reads after DS writes (in-order per wave)

    v16bf ap0 = *(const v16bf*)(lp + nl * PSE + hi * 16);
    v16bf ap1 = *(const v16bf*)(lp + nl * PSE + 32 + hi * 16);
    asm volatile("" ::: "memory");

    // ---- O += P . V : 2 key blocks x 4 dk tiles = 8 WMMAs (B from LDS)
    #pragma unroll
    for (int j = 0; j < 2; ++j) {
      const v16bf apj = j ? ap1 : ap0;
      #pragma unroll
      for (int t = 0; t < 4; ++t) {
        v16bf bv = *(const v16bf*)(lbV + (16 * t + nl) * KVS + 32 * j + hi * 16);
        acc[t] = wmma_bf16(apj, bv, acc[t]);
      }
    }
  }

  // ---- normalize + emit [b, n, h*dk] bf16 (interleaved for output GEMM)
  #pragma unroll
  for (int r = 0; r < 8; ++r) {
    const float inv = 1.0f / lrow[r];
    const int m = q0 + r + hi * 8;
    const size_t rowb = ((size_t)(b2 * NSEQ + m)) * DM + h * DK;
    #pragma unroll
    for (int t = 0; t < 4; ++t)
      attn[rowb + (t >> 1) * 32 + 2 * nl + (t & 1)] = (__bf16)(acc[t][r] * inv);
  }
}

// ---------------------------------------------------------------- launch
extern "C" void kernel_launch(void* const* d_in, const int* in_sizes, int n_in,
                              void* d_out, int out_size, void* d_ws, size_t ws_size,
                              hipStream_t stream) {
  const float* Q  = (const float*)d_in[0];
  const float* K  = (const float*)d_in[1];
  const float* V  = (const float*)d_in[2];
  const float* wq = (const float*)d_in[3];
  const float* wk = (const float*)d_in[4];
  const float* wv = (const float*)d_in[5];
  const float* wo = (const float*)d_in[6];

  char* ws = (char*)d_ws;
  const size_t WB = (size_t)DM * DM * sizeof(__bf16);
  const size_t PB = (size_t)MROWS * DM * sizeof(__bf16);
  __bf16* wqb  = (__bf16*)(ws);
  __bf16* wkb  = (__bf16*)(ws + WB);
  __bf16* wvb  = (__bf16*)(ws + 2 * WB);
  __bf16* wob  = (__bf16*)(ws + 3 * WB);
  __bf16* qpb  = (__bf16*)(ws + 4 * WB);
  __bf16* kpb  = (__bf16*)(ws + 4 * WB + PB);
  __bf16* vtb  = (__bf16*)(ws + 4 * WB + 2 * PB);
  __bf16* attn = (__bf16*)(ws + 4 * WB + 3 * PB);

  const int wn = DM * DM;
  const int cb = (wn + 255) / 256;
  cvt_k<<<cb, 256, 0, stream>>>(wq, wqb, wn, 0.125f);  // fold 1/sqrt(dk) into w_q
  cvt_k<<<cb, 256, 0, stream>>>(wk, wkb, wn, 1.0f);
  cvt_k<<<cb, 256, 0, stream>>>(wv, wvb, wn, 1.0f);
  cvt_k<<<cb, 256, 0, stream>>>(wo, wob, wn, 1.0f);

  dim3 gg(MROWS / 128, DM / 64);
  gemm_k<0><<<gg, 256, 0, stream>>>(Q, wqb, qpb);
  gemm_k<1><<<gg, 256, 0, stream>>>(K, wkb, kpb);
  gemm_k<2><<<gg, 256, 0, stream>>>(V, wvb, vtb);

  flash_k<<<dim3(NSEQ / 128, BATCH * NH), 256, 0, stream>>>(qpb, kpb, vtb, attn);

  gemm_k<3><<<gg, 256, 0, stream>>>(attn, wob, (float*)d_out);
}
